// MSDeformAttnPixelDecoderEncOnly_82068235092333
// MI455X (gfx1250) — matrix operand order repaired and checked by reading, not compiled, not run on hardware
//
#include <hip/hip_runtime.h>

typedef __attribute__((ext_vector_type(16))) _Float16 v16h;
typedef __attribute__((ext_vector_type(8)))  _Float16 v8h;
typedef __attribute__((ext_vector_type(4)))  _Float16 v4h;
typedef __attribute__((ext_vector_type(8)))  float    v8f;

#define MSDA_LEN 43054
#define MSDA_DM  256
#define MSDA_MG  86        // persistent M-groups per GEMM grid

#define SHUF16(lo, hi) __builtin_shufflevector((lo), (hi), \
    0, 1, 2, 3, 4, 5, 6, 7, 8, 9, 10, 11, 12, 13, 14, 15)

// ---------------------------------------------------------------------------
// f32 -> f16 elementwise conversion (4 elems/thread, vectorized)
// ---------------------------------------------------------------------------
__global__ __launch_bounds__(256) void k_cvt_f16(const float* __restrict__ in,
                                                 _Float16* __restrict__ out, int n) {
  int i = (blockIdx.x * 256 + threadIdx.x) * 4;
  if (i + 3 < n) {
    float4 v = *reinterpret_cast<const float4*>(in + i);
    v4h o = { (_Float16)v.x, (_Float16)v.y, (_Float16)v.z, (_Float16)v.w };
    *reinterpret_cast<v4h*>(out + i) = o;
  } else {
    for (; i < n; ++i) out[i] = (_Float16)in[i];
  }
}

// ---------------------------------------------------------------------------
// Weight transpose + f16 convert: in[K][N] f32 -> out[N][K] f16
// ---------------------------------------------------------------------------
__global__ __launch_bounds__(256) void k_transpose_cvt(const float* __restrict__ in,
                                                       _Float16* __restrict__ out,
                                                       int K, int N) {
  int i = blockIdx.x * 256 + threadIdx.x;
  if (i < K * N) {
    int k = i / N, n = i - k * N;
    out[(long)n * K + k] = (_Float16)in[i];
  }
}

// ---------------------------------------------------------------------------
// WMMA GEMM: Out[M][N] = A[M][256] * Bt[N][256]^T + bias[N]
//
// grid = (MSDA_MG, N/64). Each block owns one 64-wide column tile:
//  - Bt tile (64 x 256 f16, padded rows) staged in LDS once via
//    global_load_async_to_lds_b128 (ASYNCcnt), then read via ds_load.
//  - 8 waves each compute a 32x64 strip (2 A frags x 4 B frags = 8 WMMA per
//    k-step); block covers 256 rows per step, persistent over M.
//  - k-loop software-pipelined: next step's A (global) and B (LDS) fragments
//    are staged one iteration ahead so WMMAs hide the load latency.
// ---------------------------------------------------------------------------
template<bool OUT_F16, int N>
__global__ __launch_bounds__(256) void k_gemm_wmma(
    const _Float16* __restrict__ A,    // [M][256] row-major f16
    const _Float16* __restrict__ Bt,   // [N][256] f16 (B transposed)
    const float*    __restrict__ bias, // [N] f32
    void*           __restrict__ Out,  // [M][N] f32 or f16
    int M)
{
  constexpr int K   = 256;
  constexpr int LDB = K + 8;                 // halves per LDS row (bank pad)
  __shared__ _Float16 Bs[64 * LDB];          // 33792 bytes

  const int n0 = blockIdx.y * 64;

  // ---- async fill of the B tile: 64 rows x 256 halves = 2048 x 16B chunks ----
#pragma unroll
  for (int it = 0; it < 8; ++it) {
    const int c   = it * 256 + threadIdx.x;
    const int r   = c >> 5;
    const int c16 = c & 31;
    const unsigned long long g =
        (unsigned long long)(uintptr_t)(Bt + (size_t)(n0 + r) * K + c16 * 8);
    const unsigned int l =
        (unsigned int)(uintptr_t)(Bs + r * LDB + c16 * 8);
    asm volatile("global_load_async_to_lds_b128 %0, %1, off"
                 :: "v"(l), "v"(g) : "memory");
  }
  asm volatile("s_wait_asynccnt 0x0" ::: "memory");
  __syncthreads();

  const int lane = threadIdx.x & 31;
  const int wave = threadIdx.x >> 5;
  const int lrow = lane & 15;                // M (for A) / N-local (for B,C)
  const int lhi  = lane >> 4;                // lane-group select

  const _Float16* __restrict__ Bl = Bs + lrow * LDB + lhi * 16;

  for (long mbase = (long)blockIdx.x * 256; mbase < (long)M;
       mbase += (long)gridDim.x * 256) {
    const long m0 = mbase + wave * 32;       // 32 rows per wave
    if (m0 >= M) continue;                   // no barriers below: safe

    long ar0 = m0 + lrow;       if (ar0 >= M) ar0 = M - 1;
    long ar1 = m0 + 16 + lrow;  if (ar1 >= M) ar1 = M - 1;
    const _Float16* __restrict__ A0 = A + ar0 * K + lhi * 16;
    const _Float16* __restrict__ A1 = A + ar1 * K + lhi * 16;

    v8f c00 = {}, c01 = {}, c02 = {}, c03 = {};
    v8f c10 = {}, c11 = {}, c12 = {}, c13 = {};

    // ---- prologue: stage k0 = 0 fragments ----
    v8h a0l = *(const v8h*)(A0);
    v8h a0h = *(const v8h*)(A0 + 8);
    v8h a1l = *(const v8h*)(A1);
    v8h a1h = *(const v8h*)(A1 + 8);
    v8h b0l = *(const v8h*)(Bl);
    v8h b0h = *(const v8h*)(Bl + 8);
    v8h b1l = *(const v8h*)(Bl + 16 * LDB);
    v8h b1h = *(const v8h*)(Bl + 16 * LDB + 8);
    v8h b2l = *(const v8h*)(Bl + 32 * LDB);
    v8h b2h = *(const v8h*)(Bl + 32 * LDB + 8);
    v8h b3l = *(const v8h*)(Bl + 48 * LDB);
    v8h b3h = *(const v8h*)(Bl + 48 * LDB + 8);

#pragma unroll
    for (int k0 = 0; k0 < K; k0 += 32) {
      const v16h a0 = SHUF16(a0l, a0h);
      const v16h a1 = SHUF16(a1l, a1h);
      const v16h b0 = SHUF16(b0l, b0h);
      const v16h b1 = SHUF16(b1l, b1h);
      const v16h b2 = SHUF16(b2l, b2h);
      const v16h b3 = SHUF16(b3l, b3h);

      if (k0 + 32 < K) {                     // stage next k-step's fragments
        const int kn = k0 + 32;
        a0l = *(const v8h*)(A0 + kn);
        a0h = *(const v8h*)(A0 + kn + 8);
        a1l = *(const v8h*)(A1 + kn);
        a1h = *(const v8h*)(A1 + kn + 8);
        b0l = *(const v8h*)(Bl + kn);
        b0h = *(const v8h*)(Bl + kn + 8);
        b1l = *(const v8h*)(Bl + kn + 16 * LDB);
        b1h = *(const v8h*)(Bl + kn + 16 * LDB + 8);
        b2l = *(const v8h*)(Bl + kn + 32 * LDB);
        b2h = *(const v8h*)(Bl + kn + 32 * LDB + 8);
        b3l = *(const v8h*)(Bl + kn + 48 * LDB);
        b3h = *(const v8h*)(Bl + kn + 48 * LDB + 8);
      }

      c00 = __builtin_amdgcn_wmma_f32_16x16x32_f16(false, a0, false, b0, (short)0, c00, false, false);
      c01 = __builtin_amdgcn_wmma_f32_16x16x32_f16(false, a0, false, b1, (short)0, c01, false, false);
      c02 = __builtin_amdgcn_wmma_f32_16x16x32_f16(false, a0, false, b2, (short)0, c02, false, false);
      c03 = __builtin_amdgcn_wmma_f32_16x16x32_f16(false, a0, false, b3, (short)0, c03, false, false);
      c10 = __builtin_amdgcn_wmma_f32_16x16x32_f16(false, a1, false, b0, (short)0, c10, false, false);
      c11 = __builtin_amdgcn_wmma_f32_16x16x32_f16(false, a1, false, b1, (short)0, c11, false, false);
      c12 = __builtin_amdgcn_wmma_f32_16x16x32_f16(false, a1, false, b2, (short)0, c12, false, false);
      c13 = __builtin_amdgcn_wmma_f32_16x16x32_f16(false, a1, false, b3, (short)0, c13, false, false);
    }

    v8f cc[2][4] = { { c00, c01, c02, c03 }, { c10, c11, c12, c13 } };
    const int mb = lhi * 8;
#pragma unroll
    for (int s = 0; s < 2; ++s) {
#pragma unroll
      for (int t = 0; t < 4; ++t) {
        const int n = n0 + t * 16 + lrow;
        const float bn = bias[n];
#pragma unroll
        for (int r = 0; r < 8; ++r) {
          const long row = m0 + s * 16 + mb + r;
          if (row < M) {
            const float v = cc[s][t][r] + bn;
            if (OUT_F16) reinterpret_cast<_Float16*>(Out)[row * (long)N + n] = (_Float16)v;
            else         reinterpret_cast<float*>(Out)[row * (long)N + n] = v;
          }
        }
      }
    }
  }
}

// ---------------------------------------------------------------------------
// Deformable sampling: one wave per (query, head); lane = channel (0..31).
// Softmax over 16 logits via wave32 shfl_xor; 16 bilinear gathers from the
// f16 value tensor (22 MB -> L2 resident); result written f16 for final GEMM.
// ---------------------------------------------------------------------------
__global__ __launch_bounds__(256) void k_msda_sample(
    const float*    __restrict__ attn,  // [LEN][128] logits
    const float*    __restrict__ offs,  // [LEN][256] sampling offsets
    const float*    __restrict__ refp,  // [LEN][4][2] reference points
    const _Float16* __restrict__ val,   // [LEN][256] projected value (f16)
    _Float16*       __restrict__ outp)  // [LEN][256] per-head outputs
{
  const int q    = blockIdx.x;
  const int h    = threadIdx.x >> 5;
  const int lane = threadIdx.x & 31;

  float lg = attn[(long)q * 128 + h * 16 + (lane & 15)];
  float mx = lg;
#pragma unroll
  for (int w = 1; w < 16; w <<= 1) mx = fmaxf(mx, __shfl_xor(mx, w, 32));
  float e = __expf(lg - mx);
  float s = e;
#pragma unroll
  for (int w = 1; w < 16; w <<= 1) s += __shfl_xor(s, w, 32);
  const float invs = 1.0f / s;

  const int HW[4] = { 180, 90, 45, 23 };
  const int ST[4] = { 0, 32400, 40500, 42525 };

  const float* __restrict__ offq = offs + (long)q * 256 + h * 32;
  const float* __restrict__ refq = refp + (long)q * 8;

  float acc = 0.0f;
#pragma unroll
  for (int lvl = 0; lvl < 4; ++lvl) {
    const int   Wi = HW[lvl];
    const float Wf = (float)Wi;
    const float rx = refq[lvl * 2 + 0] * Wf - 0.5f;
    const float ry = refq[lvl * 2 + 1] * Wf - 0.5f;
    const _Float16* __restrict__ vb = val + (long)ST[lvl] * 256 + h * 32 + lane;
#pragma unroll
    for (int p = 0; p < 4; ++p) {
      const float x = rx + offq[lvl * 8 + p * 2 + 0];
      const float y = ry + offq[lvl * 8 + p * 2 + 1];
      const float xf = floorf(x), yf = floorf(y);
      const float dx = x - xf,   dy = y - yf;
      const int   x0 = (int)xf,  y0 = (int)yf;

      const int x0c = min(max(x0, 0), Wi - 1);
      const int x1c = min(max(x0 + 1, 0), Wi - 1);
      const int y0c = min(max(y0, 0), Wi - 1);
      const int y1c = min(max(y0 + 1, 0), Wi - 1);
      const float vx0 = (x0 >= 0 && x0 < Wi) ? 1.0f : 0.0f;
      const float vx1 = (x0 + 1 >= 0 && x0 + 1 < Wi) ? 1.0f : 0.0f;
      const float vy0 = (y0 >= 0 && y0 < Wi) ? 1.0f : 0.0f;
      const float vy1 = (y0 + 1 >= 0 && y0 + 1 < Wi) ? 1.0f : 0.0f;

      const float g00 = (float)vb[((long)y0c * Wi + x0c) * 256];
      const float g10 = (float)vb[((long)y0c * Wi + x1c) * 256];
      const float g01 = (float)vb[((long)y1c * Wi + x0c) * 256];
      const float g11 = (float)vb[((long)y1c * Wi + x1c) * 256];

      const float sv = g00 * ((1.f - dx) * (1.f - dy) * vx0 * vy0)
                     + g10 * (dx * (1.f - dy) * vx1 * vy0)
                     + g01 * ((1.f - dx) * dy * vx0 * vy1)
                     + g11 * (dx * dy * vx1 * vy1);

      const float aw = __shfl(e, lvl * 4 + p, 32) * invs;
      acc += aw * sv;
    }
  }
  outp[(long)q * 256 + h * 32 + lane] = (_Float16)acc;
}

// ---------------------------------------------------------------------------
extern "C" void kernel_launch(void* const* d_in, const int* in_sizes, int n_in,
                              void* d_out, int out_size, void* d_ws, size_t ws_size,
                              hipStream_t stream) {
  const float* query    = (const float*)d_in[0];
  const float* value_in = (const float*)d_in[1];
  const float* refp     = (const float*)d_in[2];
  const float* Wv       = (const float*)d_in[3];
  const float* bv       = (const float*)d_in[4];
  const float* Ws       = (const float*)d_in[5];
  const float* bs       = (const float*)d_in[6];
  const float* Wa       = (const float*)d_in[7];
  const float* ba       = (const float*)d_in[8];
  const float* Wo       = (const float*)d_in[9];
  const float* bo       = (const float*)d_in[10];

  char* ws = (char*)d_ws;
  size_t off = 0;
  auto take = [&](size_t bytes) -> void* {
    void* p = ws + off;
    off += (bytes + 255) & ~(size_t)255;
    return p;
  };
  const size_t NE = (size_t)MSDA_LEN * MSDA_DM;
  _Float16* qf    = (_Float16*)take(NE * 2);
  _Float16* vf    = (_Float16*)take(NE * 2);
  _Float16* valf  = (_Float16*)take(NE * 2);
  _Float16* tmpf  = (_Float16*)take(NE * 2);
  float*    offsf = (float*)take(NE * 4);
  float*    attnf = (float*)take((size_t)MSDA_LEN * 128 * 4);
  _Float16* Wvt   = (_Float16*)take(256 * 256 * 2);
  _Float16* Wst   = (_Float16*)take(256 * 256 * 2);
  _Float16* Wat   = (_Float16*)take(128 * 256 * 2);
  _Float16* Wot   = (_Float16*)take(256 * 256 * 2);

  // 1) activation + weight conversion
  const int ncvt = (int)NE;
  const int cblk = (ncvt / 4 + 255) / 256;
  k_cvt_f16<<<cblk, 256, 0, stream>>>(query, qf, ncvt);
  k_cvt_f16<<<cblk, 256, 0, stream>>>(value_in, vf, ncvt);
  k_transpose_cvt<<<(256 * 256 + 255) / 256, 256, 0, stream>>>(Wv, Wvt, 256, 256);
  k_transpose_cvt<<<(256 * 256 + 255) / 256, 256, 0, stream>>>(Ws, Wst, 256, 256);
  k_transpose_cvt<<<(256 * 128 + 255) / 256, 256, 0, stream>>>(Wa, Wat, 256, 128);
  k_transpose_cvt<<<(256 * 256 + 255) / 256, 256, 0, stream>>>(Wo, Wot, 256, 256);

  // 2) input GEMMs via WMMA (LDS-staged weights, async fill, persistent blocks)
  dim3 g256(MSDA_MG, 4), g128(MSDA_MG, 2);
  k_gemm_wmma<true, 256><<<g256, 256, 0, stream>>>(vf, Wvt, bv, valf, MSDA_LEN);
  k_gemm_wmma<false, 256><<<g256, 256, 0, stream>>>(qf, Wst, bs, offsf, MSDA_LEN);
  k_gemm_wmma<false, 128><<<g128, 256, 0, stream>>>(qf, Wat, ba, attnf, MSDA_LEN);

  // 3) softmax + deformable bilinear sampling (L2-resident f16 gathers)
  k_msda_sample<<<MSDA_LEN, 256, 0, stream>>>(attnf, offsf, refp, valf, tmpf);

  // 4) output projection GEMM -> d_out (f32)
  k_gemm_wmma<false, 256><<<g256, 256, 0, stream>>>(tmpf, Wot, bo, d_out, MSDA_LEN);
}
